// MADpredictor_6399501271320
// MI455X (gfx1250) — compile-verified, matrix operand order checked by reading, not run on hardware
//
#include <hip/hip_runtime.h>
#include <hip/hip_bf16.h>

#define N_HEADS    4
#define N_NODES    10000
#define EMB        128
#define N_BATCH    4096
#define KKEEP      9            // top-(k+1), drop self

typedef __attribute__((ext_vector_type(16))) __bf16 v16bf;
typedef __attribute__((ext_vector_type(8)))  float  v8f;
typedef __attribute__((ext_vector_type(4)))  int    v4i;

#define AS1 __attribute__((address_space(1)))
#define AS3 __attribute__((address_space(3)))

#if defined(__has_builtin)
#if __has_builtin(__builtin_amdgcn_global_load_async_to_lds_b128)
#define HAVE_ASYNC_LDS 1
#endif
#if __has_builtin(__builtin_amdgcn_s_wait_asynccnt)
#define HAVE_WAIT_ASYNC 1
#endif
#endif

union Frag {
    v16bf    v;
    unsigned u[8];              // 8 dwords = 16 bf16
};

// ---------------------------------------------------------------------------
// Stage 1: f32 -> bf16 (round-to-nearest-even) + per-node squared norms
// grid = N_HEADS*N_NODES blocks, 128 threads (one per dim)
// ---------------------------------------------------------------------------
__global__ __launch_bounds__(128) void convert_kernel(const float* __restrict__ embeds,
                                                      unsigned short* __restrict__ ebf,
                                                      float* __restrict__ y2)
{
    const int row = blockIdx.x;           // h*N_NODES + n
    const int d   = threadIdx.x;          // 0..127
    const size_t off = (size_t)row * EMB + d;

    float v = embeds[off];
    unsigned bits = __float_as_uint(v);
    unsigned r = (bits + 0x7FFFu + ((bits >> 16) & 1u)) >> 16;   // RNE
    ebf[off] = (unsigned short)r;

    float bv = __uint_as_float(r << 16);
    __shared__ float red[128];
    red[d] = bv * bv;
    __syncthreads();
    #pragma unroll
    for (int s = 64; s > 0; s >>= 1) {
        if (d < s) red[d] += red[d + s];
        __syncthreads();
    }
    if (d == 0) y2[row] = red[0];
}

// ---------------------------------------------------------------------------
// Stage 2: fused bf16-WMMA score GEMM + streaming top-9 per batch row.
// grid = (N_BATCH/16, N_HEADS, 2 builds), block = 128 threads (4 waves).
// Ranking key: y2[n] - 2 * <x,y>   (x^2 is row-constant, drops out of argsort)
// ---------------------------------------------------------------------------
__global__ __launch_bounds__(128) void knn_kernel(const unsigned short* __restrict__ ebf,
                                                  const float* __restrict__ y2,
                                                  const int* __restrict__ batch_edges,
                                                  int* __restrict__ samples /*[2][4][N_BATCH][8]*/)
{
    const int tileB = blockIdx.x;         // batch tile of 16 rows
    const int h     = blockIdx.y;         // head
    const int build = blockIdx.z;         // 0 = src build, 1 = dst build
    const int tid   = threadIdx.x;
    const int wave  = tid >> 5;
    const int lane  = tid & 31;
    const int half  = lane >> 4;          // 0 | 1
    const int l16   = lane & 15;

    const int* nodes = batch_edges + build * N_BATCH;
    const unsigned* eb = (const unsigned*)(ebf + (size_t)h * N_NODES * EMB); // dword view
    const float* y2h = y2 + h * N_NODES;

    __shared__ unsigned aT[16][64];       // A tile: 16 rows x 128 bf16 (4 KB)
    __shared__ float    sc[4][16][16];    // per-wave score tile
    __shared__ float    mD[4][16][KKEEP];
    __shared__ int      mI[4][16][KKEEP];

    // ---- cooperative A-tile load: 16 rows x 256 B ----
#if defined(HAVE_ASYNC_LDS)
    // CDNA5 async direct global->LDS copy (ASYNCcnt), 16 B per lane-op
    for (int e = tid; e < 256; e += 128) {
        int r = e >> 4, c = e & 15;                       // 16-byte chunk in row
        int node = nodes[tileB * 16 + r];
        void* gp = (void*)(eb + (size_t)node * 64 + c * 4);
        __builtin_amdgcn_global_load_async_to_lds_b128(
            (AS1 v4i*)gp, (AS3 v4i*)&aT[r][c * 4], 0, 0);
    }
#if defined(HAVE_WAIT_ASYNC)
    __builtin_amdgcn_s_wait_asynccnt(0);
#else
    asm volatile("s_wait_asynccnt 0" ::: "memory");
#endif
#else
    for (int i = tid; i < 16 * 64; i += 128) {
        int r = i >> 6, c = i & 63;
        int node = nodes[tileB * 16 + r];
        aT[r][c] = eb[(size_t)node * 64 + c];
    }
#endif
    __syncthreads();

    // A fragments (ISA 16-bit A layout: lane=row, K pairs striped over VGPRs,
    // half-wave carries K+8 within each 16-wide K group)
    unsigned au[4][8];
    #pragma unroll
    for (int ks = 0; ks < 4; ++ks) {
        #pragma unroll
        for (int v = 0; v < 8; ++v) {
            int k = ks * 32 + (v < 4 ? 2 * v : 2 * v + 8) + 8 * half;  // even
            au[ks][v] = aT[l16][k >> 1];
        }
    }

    float topD[KKEEP];
    int   topI[KKEEP];
    #pragma unroll
    for (int i = 0; i < KKEEP; ++i) { topD[i] = 3.0e38f; topI[i] = -1; }

    // stream all node tiles; waves take disjoint tiles (no block sync inside)
    for (int t = wave; t < N_NODES / 16; t += 4) {
        const int bn = t * 16 + l16;                       // B column = node
        const unsigned* bp = eb + (size_t)bn * 64;

        // prefetch this wave's next tile (t+4): row is 256 B = 2 cachelines
        if (t + 4 < N_NODES / 16) {
            const unsigned* np = eb + (size_t)(bn + 64) * 64;  // (t+4)*16 + l16
            __builtin_prefetch(np, 0, 3);
            __builtin_prefetch(np + 32, 0, 3);
        }

        v8f acc = {0.f, 0.f, 0.f, 0.f, 0.f, 0.f, 0.f, 0.f};
        #pragma unroll
        for (int ks = 0; ks < 4; ++ks) {
            Frag af, bfr;
            #pragma unroll
            for (int v = 0; v < 8; ++v) af.u[v] = au[ks][v];
            #pragma unroll
            for (int v = 0; v < 8; ++v) {
                int k = ks * 32 + (v < 4 ? 2 * v : 2 * v + 8) + 8 * half;
                bfr.u[v] = bp[k >> 1];
            }
            acc = __builtin_amdgcn_wmma_f32_16x16x32_bf16(
                false, af.v, false, bfr.v, (short)0, acc, false, false);
        }

        // key tile -> LDS (C layout: lane%16 = N, vgpr idx + 8*half = M)
        const float yv = y2h[bn];
        #pragma unroll
        for (int v = 0; v < 8; ++v)
            sc[wave][v + 8 * half][l16] = yv - 2.0f * acc[v];

        asm volatile("s_wait_dscnt 0" ::: "memory");       // wave-private LDS sync
        __builtin_amdgcn_wave_barrier();

        if (lane < 16) {                                    // owner lane = row
            #pragma unroll
            for (int n = 0; n < 16; ++n) {
                float key = sc[wave][lane][n];
                if (key < topD[KKEEP - 1]) {
                    int idx = t * 16 + n;
                    int p = KKEEP - 1;
                    while (p > 0 && topD[p - 1] > key) {
                        topD[p] = topD[p - 1]; topI[p] = topI[p - 1]; --p;
                    }
                    topD[p] = key; topI[p] = idx;
                }
            }
        }
        __builtin_amdgcn_wave_barrier();
    }

    if (lane < 16) {
        #pragma unroll
        for (int i = 0; i < KKEEP; ++i) { mD[wave][lane][i] = topD[i]; mI[wave][lane][i] = topI[i]; }
    }
    __syncthreads();

    // merge 4 sorted lists, drop the minimum (self match), emit 8 neighbors
    if (wave == 0 && lane < 16) {
        float bd[KKEEP]; int bi[KKEEP];
        #pragma unroll
        for (int i = 0; i < KKEEP; ++i) { bd[i] = 3.0e38f; bi[i] = -1; }
        for (int w = 0; w < 4; ++w) {
            for (int i = 0; i < KKEEP; ++i) {
                float key = mD[w][lane][i];
                if (key >= bd[KKEEP - 1]) break;           // source list is sorted
                int idx = mI[w][lane][i];
                int p = KKEEP - 1;
                while (p > 0 && bd[p - 1] > key) {
                    bd[p] = bd[p - 1]; bi[p] = bi[p - 1]; --p;
                }
                bd[p] = key; bi[p] = idx;
            }
        }
        const size_t outBase =
            ((((size_t)build * N_HEADS + h) * N_BATCH) + tileB * 16 + lane) * 8;
        #pragma unroll
        for (int j = 0; j < 8; ++j) samples[outBase + j] = bi[j + 1];
    }
}

// ---------------------------------------------------------------------------
// Stage 3: exact-f32 epilogue. grid = N_BATCH, block = 128 (wave = head).
// ---------------------------------------------------------------------------
__global__ __launch_bounds__(128) void score_kernel(const float* __restrict__ embeds,
                                                    const float* __restrict__ field,
                                                    const float* __restrict__ unc,
                                                    const float* __restrict__ adj,
                                                    const int* __restrict__ batch_edges,
                                                    const int* __restrict__ samples,
                                                    float* __restrict__ out)
{
    const int b    = blockIdx.x;
    const int h    = threadIdx.x >> 5;
    const int lane = threadIdx.x & 31;

    const int srcb = batch_edges[b];
    const int dstb = batch_edges[N_BATCH + b];
    const float* eh = embeds + (size_t)h * N_NODES * EMB;
    const float* fh = field  + (size_t)h * N_NODES * EMB;
    const float  U  = unc[0];

    float logit[16], dist[16];
    #pragma unroll
    for (int j = 0; j < 16; ++j) {
        const int build = j >> 3;
        const int nb = build ? dstb : srcb;     // node whose diffs we take
        const int ng = build ? srcb : dstb;     // node providing field g
        const int s  = samples[((((size_t)build * N_HEADS + h) * N_BATCH) + b) * 8 + (j & 7)];

        const float4* pe = (const float4*)(eh + (size_t)nb * EMB);
        const float4* ps = (const float4*)(eh + (size_t)s  * EMB);
        const float4* pg = (const float4*)(fh + (size_t)ng * EMB);
        float4 e4 = pe[lane], s4 = ps[lane], g4 = pg[lane];
        float dx = e4.x - s4.x, dy = e4.y - s4.y, dz = e4.z - s4.z, dw = e4.w - s4.w;
        float lg = dx * g4.x + dy * g4.y + dz * g4.z + dw * g4.w;
        float d2 = dx * dx + dy * dy + dz * dz + dw * dw;
        #pragma unroll
        for (int m = 16; m >= 1; m >>= 1) {
            lg += __shfl_xor(lg, m, 32);
            d2 += __shfl_xor(d2, m, 32);
        }
        float aval = build ? adj[(size_t)srcb * N_NODES + s]
                           : adj[(size_t)s * N_NODES + dstb];
        logit[j] = lg + U * (2.0f * aval - 1.0f);
        dist[j]  = sqrtf(d2);
    }

    __shared__ float hres[N_HEADS];
    if (lane == 0) {
        float num = 0.f, den = 8.0f;            // 8 sentinels: logit 0, exp(1-1)=1
        #pragma unroll
        for (int j = 0; j < 16; ++j) {
            float w = __expf(1.0f - dist[j]);
            num += logit[j] * w;
            den += w;
        }
        hres[h] = num / den;
    }
    __syncthreads();
    if (threadIdx.x == 0) {
        float m = 0.25f * (hres[0] + hres[1] + hres[2] + hres[3]);
        out[b] = 1.0f / (1.0f + __expf(-m));
    }
}

// ---------------------------------------------------------------------------
extern "C" void kernel_launch(void* const* d_in, const int* in_sizes, int n_in,
                              void* d_out, int out_size, void* d_ws, size_t ws_size,
                              hipStream_t stream)
{
    const float* embeds      = (const float*)d_in[0];
    const float* field       = (const float*)d_in[1];
    const float* uncertainty = (const float*)d_in[2];
    const float* adj         = (const float*)d_in[3];
    const int*   batch_edges = (const int*)d_in[4];
    float*       out         = (float*)d_out;

    char* ws = (char*)d_ws;
    unsigned short* ebf = (unsigned short*)ws;                       // 10,240,000 B
    float* y2  = (float*)(ws + (size_t)N_HEADS * N_NODES * EMB * 2); //    160,000 B
    int* samples = (int*)(ws + (size_t)N_HEADS * N_NODES * EMB * 2
                             + (size_t)N_HEADS * N_NODES * 4);       //  1,048,576 B

    convert_kernel<<<N_HEADS * N_NODES, 128, 0, stream>>>(embeds, ebf, y2);

    dim3 g2(N_BATCH / 16, N_HEADS, 2);
    knn_kernel<<<g2, 128, 0, stream>>>(ebf, y2, batch_edges, samples);

    score_kernel<<<N_BATCH, 128, 0, stream>>>(embeds, field, uncertainty, adj,
                                              batch_edges, samples, out);
}
